// SLatGaussianDecoder_18743237280117
// MI455X (gfx1250) — matrix operand order repaired and checked by reading, not compiled, not run on hardware
//
#include <hip/hip_runtime.h>
#include <math.h>

// ---------------------------------------------------------------------------
// SLat Gaussian Decoder for MI455X (gfx1250, wave32, WMMA)
// All GEMMs run on v_wmma_f32_16x16x32_f16 with fp32 accumulation.
// f16 weights (~171MB) fit in the 192MB L2 -> compute bound on matrix pipes.
// Wave tile 32x64 (2x4 WMMA accumulators) => ~21 flop/byte from L2.
// ---------------------------------------------------------------------------

typedef _Float16 half_t;
typedef __attribute__((ext_vector_type(16))) _Float16 v16h;
typedef __attribute__((ext_vector_type(8)))  float    v8f;
typedef __attribute__((ext_vector_type(4)))  unsigned int v4u;

union Frag16 { v16h h; v4u u[2]; };

__device__ __forceinline__ v4u ld128(const half_t* p) {
    return *(const v4u*)p;
}

// exact GELU
__device__ __forceinline__ float gelu_exact(float x) {
    return 0.5f * x * (1.0f + erff(x * 0.70710678118654752f));
}

// ---------------------------------------------------------------------------
// Embedding: h[n,c] = feats[n,:8] @ in_w[:,c] + in_b[c] + pos_embed(coords)[c]
// ---------------------------------------------------------------------------
__global__ __launch_bounds__(256) void embed_kernel(
    const float* __restrict__ feats, const int* __restrict__ coords,
    const float* __restrict__ in_w, const float* __restrict__ in_b,
    float* __restrict__ h, int n)
{
    int t = blockIdx.x * 256 + threadIdx.x;
    if (t >= n * 768) return;
    int c = t % 768;
    int i = t / 768;
    float acc = in_b[c];
    #pragma unroll
    for (int j = 0; j < 8; ++j)
        acc += feats[i * 8 + j] * in_w[j * 768 + c];
    // positional embedding: axis a = c/256, r = c%256, f = r&127
    int a = c / 256;
    int r = c % 256;
    int f = r & 127;
    // freq = 10000^(-f/128) = exp(-f * ln(10000)/128)
    float freq = __expf(-(float)f * (9.210340371976184f / 128.0f));
    float phase = (float)coords[i * 3 + a] * freq;
    acc += (r < 128) ? __sinf(phase) : __cosf(phase);
    h[t] = acc;
}

// ---------------------------------------------------------------------------
// Weight convert + transpose: W[K,N] f32 -> Wt[N,K] f16 (K-contiguous rows)
// ---------------------------------------------------------------------------
__global__ __launch_bounds__(256) void convert_wt_kernel(
    const float* __restrict__ W, half_t* __restrict__ Wt, int K, int Nn)
{
    int t = blockIdx.x * 256 + threadIdx.x;
    if (t >= K * Nn) return;
    int k = t / Nn;
    int nn = t % Nn;
    Wt[(size_t)nn * K + k] = (half_t)W[(size_t)k * Nn + nn];
}

// ---------------------------------------------------------------------------
// LayerNorm (per row over C) -> f16 staging buffer for the next GEMM
// one wave per row; C = 768 -> 24 elements per lane
// ---------------------------------------------------------------------------
__global__ __launch_bounds__(256) void ln_f16_kernel(
    const float* __restrict__ x, half_t* __restrict__ y, int Cc, float eps)
{
    int wave = threadIdx.x >> 5;
    int lane = threadIdx.x & 31;
    int row  = blockIdx.x * 8 + wave;
    const float* xr = x + (size_t)row * Cc;
    float s = 0.0f, s2 = 0.0f;
    for (int c = lane; c < Cc; c += 32) {
        float v = xr[c];
        s += v; s2 += v * v;
    }
    #pragma unroll
    for (int off = 16; off > 0; off >>= 1) {
        s  += __shfl_xor(s,  off, 32);
        s2 += __shfl_xor(s2, off, 32);
    }
    float mu   = s / (float)Cc;
    float var  = s2 / (float)Cc - mu * mu;
    float rinv = rsqrtf(var + eps);
    half_t* yr = y + (size_t)row * Cc;
    for (int c = lane; c < Cc; c += 32)
        yr[c] = (half_t)((xr[c] - mu) * rinv);
}

// ---------------------------------------------------------------------------
// WMMA GEMM: Out[M,Nn] = act(A[M,K]_f16 @ Wt[Nn,K]_f16^T + bias)
// MODE 0: store fp32        MODE 1: fp32 residual add        MODE 2: GELU->f16
// Block = 256 threads = 8 waves arranged 4(M) x 2(N); block tile 128x128.
// Each wave computes a 32x64 tile via 2x4 v_wmma_f32_16x16x32_f16 per K step.
// ---------------------------------------------------------------------------
template <int MODE>
__global__ __launch_bounds__(256) void gemm_wmma_kernel(
    const half_t* __restrict__ A,   // [M,K] row-major f16
    const half_t* __restrict__ Wt,  // [Nn,K] row-major f16 (transposed weight)
    const float*  __restrict__ bias,// [Nn]
    float*        __restrict__ Cf,  // fp32 out (MODE 0 store / MODE 1 +=)
    half_t*       __restrict__ Ch,  // f16 out (MODE 2)
    int M, int Nn, int K)
{
    int wave = threadIdx.x >> 5;
    int lane = threadIdx.x & 31;
    int wm = wave >> 1;                        // 0..3 along M
    int wn = wave & 1;                         // 0..1 along N
    int tileM = blockIdx.y * 128 + wm * 32;
    int tileN = blockIdx.x * 128 + wn * 64;
    if (tileM >= M || tileN >= Nn) return;     // wave-uniform (Nn % 64 == 0)

    int r  = lane & 15;
    int hi = lane >> 4;                        // 0 or 1 -> K half selector

    // A fragment rows (this code's contiguous k order: lanes 0-15 K 0..15,
    // lanes 16-31 K 16..31)
    const half_t* arow0 = A + (size_t)(tileM + r)      * K;
    const half_t* arow1 = A + (size_t)(tileM + 16 + r) * K;
    // B fragment rows (same contiguous k order as A: lanes 0-15 K 0..15, lanes 16-31 K 16..31)
    const half_t* brow0 = Wt + (size_t)(tileN +  0 + r) * K + hi * 16;
    const half_t* brow1 = Wt + (size_t)(tileN + 16 + r) * K + hi * 16;
    const half_t* brow2 = Wt + (size_t)(tileN + 32 + r) * K + hi * 16;
    const half_t* brow3 = Wt + (size_t)(tileN + 48 + r) * K + hi * 16;

    v8f acc[2][4] = {};

    for (int k0 = 0; k0 < K; k0 += 32) {
        __builtin_prefetch(arow0 + k0 + 256, 0, 3);   // WGP-scope prefetch
        __builtin_prefetch(brow0 + k0 + 256, 0, 3);

        Frag16 a0, a1, b0, b1, b2, b3;
        a0.u[0] = ld128(arow0 + k0 + hi * 16);
        a0.u[1] = ld128(arow0 + k0 + hi * 16 + 8);
        a1.u[0] = ld128(arow1 + k0 + hi * 16);
        a1.u[1] = ld128(arow1 + k0 + hi * 16 + 8);
        b0.u[0] = ld128(brow0 + k0);
        b0.u[1] = ld128(brow0 + k0 + 8);
        b1.u[0] = ld128(brow1 + k0);
        b1.u[1] = ld128(brow1 + k0 + 8);
        b2.u[0] = ld128(brow2 + k0);
        b2.u[1] = ld128(brow2 + k0 + 8);
        b3.u[0] = ld128(brow3 + k0);
        b3.u[1] = ld128(brow3 + k0 + 8);

        acc[0][0] = __builtin_amdgcn_wmma_f32_16x16x32_f16(false, a0.h, false, b0.h,
                                                           (short)0, acc[0][0], false, false);
        acc[0][1] = __builtin_amdgcn_wmma_f32_16x16x32_f16(false, a0.h, false, b1.h,
                                                           (short)0, acc[0][1], false, false);
        acc[0][2] = __builtin_amdgcn_wmma_f32_16x16x32_f16(false, a0.h, false, b2.h,
                                                           (short)0, acc[0][2], false, false);
        acc[0][3] = __builtin_amdgcn_wmma_f32_16x16x32_f16(false, a0.h, false, b3.h,
                                                           (short)0, acc[0][3], false, false);
        acc[1][0] = __builtin_amdgcn_wmma_f32_16x16x32_f16(false, a1.h, false, b0.h,
                                                           (short)0, acc[1][0], false, false);
        acc[1][1] = __builtin_amdgcn_wmma_f32_16x16x32_f16(false, a1.h, false, b1.h,
                                                           (short)0, acc[1][1], false, false);
        acc[1][2] = __builtin_amdgcn_wmma_f32_16x16x32_f16(false, a1.h, false, b2.h,
                                                           (short)0, acc[1][2], false, false);
        acc[1][3] = __builtin_amdgcn_wmma_f32_16x16x32_f16(false, a1.h, false, b3.h,
                                                           (short)0, acc[1][3], false, false);
    }

    // Epilogue.  C/D 16x16 f32 layout: VGPR rr: lanes 0-15 M=rr, lanes 16-31 M=rr+8
    int moff = hi * 8;
    #pragma unroll
    for (int i = 0; i < 2; ++i) {
        #pragma unroll
        for (int j = 0; j < 4; ++j) {
            int nbase = tileN + j * 16 + r;
            if (nbase >= Nn) continue;
            float bv = bias[nbase];
            #pragma unroll
            for (int rr = 0; rr < 8; ++rr) {
                int row = tileM + i * 16 + moff + rr;
                float v = acc[i][j][rr] + bv;
                size_t o = (size_t)row * Nn + nbase;
                if (MODE == 0) {
                    Cf[o] = v;
                } else if (MODE == 1) {
                    Cf[o] += v;
                } else {
                    Ch[o] = (half_t)gelu_exact(v);
                }
            }
        }
    }
}

// ---------------------------------------------------------------------------
// Window capacity: cap = max(vslot) + 1 (data-dependent; computed on device)
// ---------------------------------------------------------------------------
__global__ void init_caps_kernel(int* caps) {
    if (threadIdx.x < 2 && blockIdx.x == 0) caps[threadIdx.x] = 0;
}
__global__ __launch_bounds__(256) void max_slot_kernel(
    const int* __restrict__ vslot, int* cap, int n)
{
    int i = blockIdx.x * 256 + threadIdx.x;
    if (i < n) atomicMax(cap, vslot[i] + 1);
}

// ---------------------------------------------------------------------------
// Windowed attention, one wave per (voxel, head), online softmax over the
// window members.  idx padding entries == n (so no mask array is needed).
// Since idx[vwin[i], vslot[i]] == i, writing per voxel is exactly the
// reference's ow[vwin, vslot] gather.  Output written as f16 for proj GEMM.
// ---------------------------------------------------------------------------
__global__ __launch_bounds__(32) void win_attn_kernel(
    const float* __restrict__ qkv,   // [n, 3, 12, 64]
    const int*   __restrict__ idx,   // [nw, cap]
    const int*   __restrict__ vwin,  // [n]
    const int*   __restrict__ capp,  // scalar
    half_t*      __restrict__ oh,    // [n, 768] f16
    int n)
{
    int i    = blockIdx.x;
    int h    = blockIdx.y;
    int lane = threadIdx.x;
    int cap  = *capp;
    int w    = vwin[i];

    const float* qb = qkv + (size_t)i * 2304 + h * 64;   // sel 0
    float q0 = qb[lane];
    float q1 = qb[lane + 32];

    float m = -1e30f, l = 0.0f, o0 = 0.0f, o1 = 0.0f;
    const int* row = idx + (size_t)w * cap;
    for (int s = 0; s < cap; ++s) {
        int j = row[s];
        if (j >= n) continue;                     // padding slot (uniform)
        const float* kb = qkv + (size_t)j * 2304 + 768 + h * 64;
        float part = q0 * kb[lane] + q1 * kb[lane + 32];
        #pragma unroll
        for (int off = 16; off > 0; off >>= 1)
            part += __shfl_xor(part, off, 32);
        float sc = part * 0.125f;                 // 1/sqrt(64)
        float mn = fmaxf(m, sc);
        float alpha = __expf(m - mn);
        float e     = __expf(sc - mn);
        const float* vb = qkv + (size_t)j * 2304 + 1536 + h * 64;
        l  = l  * alpha + e;
        o0 = o0 * alpha + e * vb[lane];
        o1 = o1 * alpha + e * vb[lane + 32];
        m = mn;
    }
    float inv = 1.0f / l;                         // window always contains self
    oh[(size_t)i * 768 + h * 64 + lane]      = (half_t)(o0 * inv);
    oh[(size_t)i * 768 + h * 64 + lane + 32] = (half_t)(o1 * inv);
}

// ---------------------------------------------------------------------------
// Gaussian decode: out[N*32, 14]
// ---------------------------------------------------------------------------
__global__ __launch_bounds__(256) void decode_kernel(
    const float* __restrict__ outp,     // [n,448]
    const int*   __restrict__ coords,   // [n,3]
    const float* __restrict__ perturb,  // [32,3]
    float* __restrict__ out, int n)
{
    int t = blockIdx.x * 256 + threadIdx.x;
    if (t >= n * 32) return;
    int g = t & 31;
    int i = t >> 5;
    const float* rowp = outp + (size_t)i * 448;
    float* orow = out + (size_t)t * 14;
    #pragma unroll
    for (int a = 0; a < 3; ++a) {
        float xyz = ((float)coords[i * 3 + a] + 0.5f) * (1.0f / 64.0f);
        float off = tanhf(rowp[g * 3 + a] + perturb[g * 3 + a]) * (0.75f / 64.0f);
        orow[a]     = xyz + off;
        orow[3 + a] = rowp[96  + g * 3 + a];
        orow[6 + a] = rowp[192 + g * 3 + a];
    }
    #pragma unroll
    for (int q = 0; q < 4; ++q)
        orow[9 + q] = rowp[288 + g * 4 + q] * 0.1f;
    orow[13] = rowp[416 + g];
}

// ---------------------------------------------------------------------------
// Host-side orchestration
// ---------------------------------------------------------------------------
extern "C" void kernel_launch(void* const* d_in, const int* in_sizes, int n_in,
                              void* d_out, int out_size, void* d_ws, size_t ws_size,
                              hipStream_t stream)
{
    const float* feats   = (const float*)d_in[0];
    const int*   coords  = (const int*)  d_in[1];
    const float* in_w    = (const float*)d_in[2];
    const float* in_b    = (const float*)d_in[3];
    const float* qkv_w   = (const float*)d_in[4];
    const float* qkv_b   = (const float*)d_in[5];
    const float* proj_w  = (const float*)d_in[6];
    const float* proj_b  = (const float*)d_in[7];
    const float* mlp_w1  = (const float*)d_in[8];
    const float* mlp_b1  = (const float*)d_in[9];
    const float* mlp_w2  = (const float*)d_in[10];
    const float* mlp_b2  = (const float*)d_in[11];
    const float* out_w   = (const float*)d_in[12];
    const float* out_b   = (const float*)d_in[13];
    const float* perturb = (const float*)d_in[14];
    const int*   idx0    = (const int*)  d_in[15];
    const int*   vwin0   = (const int*)  d_in[17];
    const int*   vslot0  = (const int*)  d_in[18];
    const int*   idx1    = (const int*)  d_in[19];
    const int*   vwin1   = (const int*)  d_in[21];
    const int*   vslot1  = (const int*)  d_in[22];

    const int N   = in_sizes[0] / 8;   // 8192
    const int C   = 768;
    const int H   = 12;
    const int NB  = 12;
    const int OUT = 448;

    // ---- workspace layout (all offsets multiples of 256 bytes) ----
    char* ws = (char*)d_ws;
    size_t off = 0;
    float* h = (float*)(ws + off);             off += (size_t)N * C * 4;      // 25.2MB
    half_t* xh = (half_t*)(ws + off);          off += (size_t)N * C * 2;      // 12.6MB
    float* qkv = (float*)(ws + off);           off += (size_t)N * 3 * C * 4;  // 75.5MB
    half_t* oh = (half_t*)(ws + off);          off += (size_t)N * C * 2;      // 12.6MB
    half_t* hid = (half_t*)(ws + off);         off += (size_t)N * 4 * C * 2;  // 50.3MB
    float* outp = (float*)(ws + off);          off += (size_t)N * OUT * 4;    // 14.7MB
    int* caps = (int*)(ws + off);              off += 256;
    half_t* qkv_wt  = (half_t*)(ws + off);     off += (size_t)NB * 3 * C * C * 2;
    half_t* proj_wt = (half_t*)(ws + off);     off += (size_t)NB * C * C * 2;
    half_t* w1t     = (half_t*)(ws + off);     off += (size_t)NB * 4 * C * C * 2;
    half_t* w2t     = (half_t*)(ws + off);     off += (size_t)NB * 4 * C * C * 2;
    half_t* out_wt  = (half_t*)(ws + off);     off += (size_t)OUT * C * 2;

    dim3 b256(256);

    // ---- embed ----
    embed_kernel<<<(N * C) / 256, b256, 0, stream>>>(feats, coords, in_w, in_b, h, N);

    // ---- weight conversion (f32 -> f16, transposed to [N,K]) ----
    auto conv = [&](const float* src, half_t* dst, int K, int Nn) {
        int total = K * Nn;
        convert_wt_kernel<<<(total + 255) / 256, b256, 0, stream>>>(src, dst, K, Nn);
    };
    for (int i = 0; i < NB; ++i) {
        conv(qkv_w  + (size_t)i * C * 3 * C, qkv_wt  + (size_t)i * 3 * C * C, C,     3 * C);
        conv(proj_w + (size_t)i * C * C,     proj_wt + (size_t)i * C * C,     C,     C);
        conv(mlp_w1 + (size_t)i * C * 4 * C, w1t     + (size_t)i * 4 * C * C, C,     4 * C);
        conv(mlp_w2 + (size_t)i * 4 * C * C, w2t     + (size_t)i * 4 * C * C, 4 * C, C);
    }
    conv(out_w, out_wt, C, OUT);

    // ---- window capacities ----
    init_caps_kernel<<<1, 32, 0, stream>>>(caps);
    max_slot_kernel<<<(N + 255) / 256, b256, 0, stream>>>(vslot0, caps + 0, N);
    max_slot_kernel<<<(N + 255) / 256, b256, 0, stream>>>(vslot1, caps + 1, N);

    // ---- transformer blocks ----
    dim3 gQKV((3 * C + 127) / 128, N / 128);   // 18 x 64
    dim3 gPRJ((C + 127) / 128,     N / 128);   //  6 x 64
    dim3 gM1 ((4 * C + 127) / 128, N / 128);   // 24 x 64
    dim3 gOUT((OUT + 127) / 128,   N / 128);   //  4 x 64 (last col-block half)
    dim3 gATT(N, H);

    for (int i = 0; i < NB; ++i) {
        const int* idx  = (i & 1) ? idx1  : idx0;
        const int* vwin = (i & 1) ? vwin1 : vwin0;
        int*       capp = (i & 1) ? caps + 1 : caps + 0;

        ln_f16_kernel<<<N / 8, b256, 0, stream>>>(h, xh, C, 1e-6f);
        gemm_wmma_kernel<0><<<gQKV, b256, 0, stream>>>(
            xh, qkv_wt + (size_t)i * 3 * C * C, qkv_b + (size_t)i * 3 * C,
            qkv, nullptr, N, 3 * C, C);
        win_attn_kernel<<<gATT, 32, 0, stream>>>(qkv, idx, vwin, capp, oh, N);
        gemm_wmma_kernel<1><<<gPRJ, b256, 0, stream>>>(
            oh, proj_wt + (size_t)i * C * C, proj_b + (size_t)i * C,
            h, nullptr, N, C, C);
        ln_f16_kernel<<<N / 8, b256, 0, stream>>>(h, xh, C, 1e-6f);
        gemm_wmma_kernel<2><<<gM1, b256, 0, stream>>>(
            xh, w1t + (size_t)i * 4 * C * C, mlp_b1 + (size_t)i * 4 * C,
            nullptr, hid, N, 4 * C, C);
        gemm_wmma_kernel<1><<<gPRJ, b256, 0, stream>>>(
            hid, w2t + (size_t)i * 4 * C * C, mlp_b2 + (size_t)i * C,
            h, nullptr, N, C, 4 * C);
    }

    // ---- head ----
    ln_f16_kernel<<<N / 8, b256, 0, stream>>>(h, xh, C, 1e-5f);
    gemm_wmma_kernel<0><<<gOUT, b256, 0, stream>>>(
        xh, out_wt, out_b, outp, nullptr, N, OUT, C);
    decode_kernel<<<(N * 32 + 255) / 256, b256, 0, stream>>>(
        outp, coords, perturb, (float*)d_out, N);
}
